// Lstm_model_78864189489522
// MI455X (gfx1250) — matrix operand order repaired and checked by reading, not compiled, not run on hardware
//
#include <hip/hip_runtime.h>
#include <hip/hip_bf16.h>

// ---------------- CDNA5 WMMA types ----------------
typedef __attribute__((ext_vector_type(16))) __bf16 v16bf;
typedef __attribute__((ext_vector_type(8)))  float  v8f;

union AFrag { v16bf v; unsigned int u[8]; __bf16 h[16]; };

#define V_  200000
#define D_  256
#define H_  512
#define T_  7
#define S_TAG 5
#define E_TAG 6
#define L_  8192
#define NEGF (-10000.0f)

#define XPAD  8
#define H1PAD 8
#define H2PAD 4

// ---- workspace layout (bytes) ----
#define OFF_BP0    0u                         // bf16 [8 kt][128 nt][32 lane][16 e]  (1 MB)
#define OFF_BP1    1048576u                   // bf16 [16][128][32][16]              (2 MB)
#define OFF_BIAS0  3145728u                   // f32 [2048]
#define OFF_BIAS1  3153920u                   // f32 [2048]
#define OFF_LOGITS 3162112u                   // f32 [8192][8]  (stride 8)
#define OFF_P      3424256u                   // f32 [64][64]   chunk (max,+) products
#define OFF_FVS    3440640u                   // f32 [64][8]    fv at chunk starts
#define OFF_BP     3442688u                   // u8  [8192][8]  backpointers
#define OFF_F      3508224u                   // u8  [64][8]    per-chunk composed maps
#define OFF_ENTRY  3508736u                   // i32 [64]       carry entering each chunk
#define OFF_BEST   3508992u                   // i32 [1]

// =====================================================================
// Prep: pack W_ih (f32, row-major [4H,K]) into the WMMA B-fragment
// layout: B element for ktile kt / ntile nt, lane l (N = l%16),
// e (K = 16*(e/8) + 8*(l/16) + e%8) is W[nt*16+l%16][kt*32+K], contiguous per lane (32B).
// =====================================================================
__global__ void pack_w0(const float* __restrict__ W, __bf16* __restrict__ Bp) {
    int d = blockIdx.x * blockDim.x + threadIdx.x;        // 0 .. 2048*256-1
    int e    = d & 15;
    int lane = (d >> 4) & 31;
    int nt   = (d >> 9) & 127;
    int kt   = d >> 16;                                   // 0..7
    Bp[d] = (__bf16)W[(nt * 16 + (lane & 15)) * D_ + kt * 32 + ((e >> 3) << 4) + ((lane >> 4) << 3) + (e & 7)];
}

__global__ void pack_w1(const float* __restrict__ W, __bf16* __restrict__ Bp) {
    int d = blockIdx.x * blockDim.x + threadIdx.x;        // 0 .. 2048*512-1
    int e    = d & 15;
    int lane = (d >> 4) & 31;
    int nt   = (d >> 9) & 127;
    int kt   = d >> 16;                                   // 0..15
    Bp[d] = (__bf16)W[(nt * 16 + (lane & 15)) * H_ + kt * 32 + ((e >> 3) << 4) + ((lane >> 4) << 3) + (e & 7)];
}

__global__ void prep_bias(const float* __restrict__ bi0, const float* __restrict__ bh0,
                          const float* __restrict__ bi1, const float* __restrict__ bh1,
                          float* __restrict__ o0, float* __restrict__ o1) {
    int t = blockIdx.x * blockDim.x + threadIdx.x;
    if (t < 4 * H_) { o0[t] = bi0[t] + bh0[t]; o1[t] = bi1[t] + bh1[t]; }
}

// =====================================================================
// Fused embed -> LSTM-step0 -> LSTM-step1 -> logits, 16 tokens per block.
// 256 threads = 8 waves; each wave owns 4 feature tiles (4x16 = 64 of
// the 512 hidden columns), computing i,g,o gate tiles (f gate skipped:
// c0 = 0 makes the forget gate a no-op).
// =====================================================================
__global__ __launch_bounds__(256)
void fused_lstm(const int* __restrict__ sent, const float* __restrict__ emb,
                const __bf16* __restrict__ Bp0, const __bf16* __restrict__ Bp1,
                const float* __restrict__ bias0, const float* __restrict__ bias1,
                const float* __restrict__ W_lin, const float* __restrict__ b_lin,
                float* __restrict__ logits) {
    __shared__ __bf16 xs [16][D_ + XPAD];
    __shared__ __bf16 h1s[16][H_ + H1PAD];
    __shared__ float  h2s[16][H_ + H2PAD];

    const int tid  = threadIdx.x;
    const int lane = tid & 31;
    const int wave = tid >> 5;
    const int row0 = blockIdx.x * 16;

    // ---- gather 16 embedding rows -> LDS (bf16) ----
    {
        int r  = tid >> 4;            // token row 0..15
        int c0 = (tid & 15) * 16;     // 16 consecutive floats per thread
        long base = (long)sent[row0 + r] * D_;
        #pragma unroll
        for (int k = 0; k < 16; ++k) xs[r][c0 + k] = (__bf16)emb[base + c0 + k];
    }
    __syncthreads();

    const int mrow = lane & 15;                  // A-matrix row (token)
    const int hiK  = (lane >> 4) ? 8 : 0;        // A-layout lane-group K offset
    const int mhi  = (lane >> 4) * 8;            // C/D-layout M offset
    const int ncol = lane & 15;                  // C/D-layout N (column in tile)

    // ---------------- layer 0: x[16,256] -> h1[16,512] ----------------
    for (int nti = 0; nti < 4; ++nti) {
        int nt = wave * 4 + nti;                 // hidden-column tile 0..31
        v8f acc_i = {}, acc_g = {}, acc_o = {};
        #pragma unroll
        for (int kt = 0; kt < D_ / 32; ++kt) {
            AFrag a;
            #pragma unroll
            for (int v = 0; v < 8; ++v) {
                int kk = kt * 32 + ((v >= 4) ? 16 : 0) + hiK + 2 * (v & 3);
                a.u[v] = *(const unsigned int*)&xs[mrow][kk];
            }
            const v16bf bi = *(const v16bf*)(Bp0 + ((size_t)(kt * 128 + (0  + nt)) * 32 + lane) * 16);
            const v16bf bg = *(const v16bf*)(Bp0 + ((size_t)(kt * 128 + (64 + nt)) * 32 + lane) * 16);
            const v16bf bo = *(const v16bf*)(Bp0 + ((size_t)(kt * 128 + (96 + nt)) * 32 + lane) * 16);
            acc_i = __builtin_amdgcn_wmma_f32_16x16x32_bf16(false, a.v, false, bi, (short)0, acc_i, false, false);
            acc_g = __builtin_amdgcn_wmma_f32_16x16x32_bf16(false, a.v, false, bg, (short)0, acc_g, false, false);
            acc_o = __builtin_amdgcn_wmma_f32_16x16x32_bf16(false, a.v, false, bo, (short)0, acc_o, false, false);
        }
        float b_i = bias0[0    + nt * 16 + ncol];
        float b_g = bias0[1024 + nt * 16 + ncol];
        float b_o = bias0[1536 + nt * 16 + ncol];
        #pragma unroll
        for (int r = 0; r < 8; ++r) {
            float gi = acc_i[r] + b_i;
            float gg = acc_g[r] + b_g;
            float go = acc_o[r] + b_o;
            float c  = (1.0f / (1.0f + __expf(-gi))) * tanhf(gg);
            float h  = (1.0f / (1.0f + __expf(-go))) * tanhf(c);
            h1s[mhi + r][nt * 16 + ncol] = (__bf16)h;
        }
    }
    __syncthreads();

    // ---------------- layer 1: h1[16,512] -> h2[16,512] ----------------
    for (int nti = 0; nti < 4; ++nti) {
        int nt = wave * 4 + nti;
        v8f acc_i = {}, acc_g = {}, acc_o = {};
        #pragma unroll
        for (int kt = 0; kt < H_ / 32; ++kt) {
            AFrag a;
            #pragma unroll
            for (int v = 0; v < 8; ++v) {
                int kk = kt * 32 + ((v >= 4) ? 16 : 0) + hiK + 2 * (v & 3);
                a.u[v] = *(const unsigned int*)&h1s[mrow][kk];
            }
            const v16bf bi = *(const v16bf*)(Bp1 + ((size_t)(kt * 128 + (0  + nt)) * 32 + lane) * 16);
            const v16bf bg = *(const v16bf*)(Bp1 + ((size_t)(kt * 128 + (64 + nt)) * 32 + lane) * 16);
            const v16bf bo = *(const v16bf*)(Bp1 + ((size_t)(kt * 128 + (96 + nt)) * 32 + lane) * 16);
            acc_i = __builtin_amdgcn_wmma_f32_16x16x32_bf16(false, a.v, false, bi, (short)0, acc_i, false, false);
            acc_g = __builtin_amdgcn_wmma_f32_16x16x32_bf16(false, a.v, false, bg, (short)0, acc_g, false, false);
            acc_o = __builtin_amdgcn_wmma_f32_16x16x32_bf16(false, a.v, false, bo, (short)0, acc_o, false, false);
        }
        float b_i = bias1[0    + nt * 16 + ncol];
        float b_g = bias1[1024 + nt * 16 + ncol];
        float b_o = bias1[1536 + nt * 16 + ncol];
        #pragma unroll
        for (int r = 0; r < 8; ++r) {
            float gi = acc_i[r] + b_i;
            float gg = acc_g[r] + b_g;
            float go = acc_o[r] + b_o;
            float c  = (1.0f / (1.0f + __expf(-gi))) * tanhf(gg);
            float h  = (1.0f / (1.0f + __expf(-go))) * tanhf(c);
            h2s[mhi + r][nt * 16 + ncol] = h;
        }
    }
    __syncthreads();

    // ---------------- logits: h2 @ W_lin.T + b_lin (T=7, tiny) ----------------
    if (tid < 16 * T_) {
        int r = tid / T_, tg = tid % T_;
        float s = b_lin[tg];
        const float* w = W_lin + tg * H_;
        for (int k = 0; k < H_; ++k) s += h2s[r][k] * w[k];
        logits[(size_t)(row0 + r) * 8 + tg] = s;
    }
}

// =====================================================================
// Viterbi as a (max,+) matrix-semiring scan.  A_t[i][j] = trans[i][j] +
// feat_t[j].  fv_t = fv_{t-1} (x) A_t.  Lane j owns column j; cross-lane
// terms via wave32 shuffles.
// =====================================================================

// Phase 1: 64 chunks x 128 steps -> per-chunk product P_c (7x7).
__global__ __launch_bounds__(32)
void viterbi_chunk_prod(const float* __restrict__ logits, const float* __restrict__ trans,
                        float* __restrict__ Pws) {
    int c = blockIdx.x, lane = threadIdx.x;
    int j = (lane < T_) ? lane : 0;
    float tcol[T_];
    #pragma unroll
    for (int i = 0; i < T_; ++i) tcol[i] = trans[i * T_ + j];
    int t0 = c * 128;
    float P[T_];
    float f0 = logits[(size_t)t0 * 8 + j];
    #pragma unroll
    for (int i = 0; i < T_; ++i) P[i] = tcol[i] + f0;        // P = A_{t0}
    for (int t = t0 + 1; t < t0 + 128; ++t) {
        float ft = logits[(size_t)t * 8 + j];
        float Pn[T_];
        #pragma unroll
        for (int i = 0; i < T_; ++i) {
            float m = -3.0e38f;
            #pragma unroll
            for (int k = 0; k < T_; ++k)
                m = fmaxf(m, __shfl(P[i], k, 32) + tcol[k]); // (P (x) A_t)[i][j]
            Pn[i] = m + ft;
        }
        #pragma unroll
        for (int i = 0; i < T_; ++i) P[i] = Pn[i];
    }
    if (lane < T_)
        for (int i = 0; i < T_; ++i) Pws[c * 64 + i * 8 + lane] = P[i];
}

// Phase 2: sequential scan over 64 chunk products -> fv at chunk starts,
// terminal scores, best tag, path score.
__global__ __launch_bounds__(32)
void viterbi_boundary(const float* __restrict__ Pws, const float* __restrict__ trans,
                      float* __restrict__ fvs, float* __restrict__ d_out,
                      int* __restrict__ bestws) {
    int lane = threadIdx.x;
    int j = (lane < T_) ? lane : 0;
    float fv = (j == S_TAG) ? 0.0f : NEGF;
    for (int c = 0; c < 64; ++c) {
        if (lane < T_) fvs[c * 8 + j] = fv;
        float m = -3.0e38f;
        #pragma unroll
        for (int i = 0; i < T_; ++i)
            m = fmaxf(m, __shfl(fv, i, 32) + Pws[c * 64 + i * 8 + j]);
        fv = m;
    }
    float term = fv + trans[j * T_ + E_TAG];
    float bs = -3.0e38f; int bi = 0;
    #pragma unroll
    for (int i = 0; i < T_; ++i) {
        float v = __shfl(term, i, 32);
        if (v > bs) { bs = v; bi = i; }        // first-max tie-break (argmax)
    }
    if (lane == 0) { d_out[0] = bs; bestws[0] = bi; }
}

// Phase 3: replay each chunk from its start fv, emit backpointers, and
// compose the chunk's backward tag-map F_c.
__global__ __launch_bounds__(32)
void viterbi_replay(const float* __restrict__ logits, const float* __restrict__ trans,
                    const float* __restrict__ fvs, unsigned char* __restrict__ bpws,
                    unsigned char* __restrict__ Fws) {
    __shared__ unsigned char bps[128][8];
    int c = blockIdx.x, lane = threadIdx.x;
    int j = (lane < T_) ? lane : 0;
    float tcol[T_];
    #pragma unroll
    for (int i = 0; i < T_; ++i) tcol[i] = trans[i * T_ + j];
    float fv = fvs[c * 8 + j];
    int t0 = c * 128;
    for (int tt = 0; tt < 128; ++tt) {
        int t = t0 + tt;
        float m = -3.0e38f; int bi = 0;
        #pragma unroll
        for (int i = 0; i < T_; ++i) {
            float v = __shfl(fv, i, 32) + tcol[i];
            if (v > m) { m = v; bi = i; }
        }
        if (lane < T_) {
            bps[tt][j] = (unsigned char)bi;
            bpws[(size_t)t * 8 + j] = (unsigned char)bi;
        }
        fv = m + logits[(size_t)t * 8 + j];
    }
    __syncthreads();
    int mtag = j;                               // backward map composition
    for (int tt = 127; tt >= 0; --tt) mtag = bps[tt][mtag];
    if (lane < T_) Fws[c * 8 + j] = (unsigned char)mtag;
}

// Phase 4: scan composed maps back-to-front -> carry entering each chunk.
__global__ void viterbi_entry(const unsigned char* __restrict__ Fws,
                              const int* __restrict__ bestws, int* __restrict__ entryws) {
    if (threadIdx.x == 0) {
        int carry = bestws[0];
        for (int c = 63; c >= 0; --c) { entryws[c] = carry; carry = Fws[c * 8 + carry]; }
    }
}

// Phase 5: 64 parallel backtraces of 128 steps each (bp staged in LDS).
__global__ __launch_bounds__(32)
void viterbi_backtrace(const unsigned char* __restrict__ bpws,
                       const int* __restrict__ entryws, float* __restrict__ d_out) {
    __shared__ unsigned char bps[128][8];
    int c = blockIdx.x, t0 = c * 128;
    const unsigned int* src = (const unsigned int*)(bpws + (size_t)t0 * 8);
    unsigned int* dst = (unsigned int*)&bps[0][0];
    for (int i = threadIdx.x; i < 256; i += 32) dst[i] = src[i];
    __syncthreads();
    if (threadIdx.x == 0) {
        int carry = entryws[c];
        for (int tt = 127; tt >= 0; --tt) {
            d_out[1 + t0 + tt] = (float)carry;  // path[t] = carry before step t
            carry = bps[tt][carry];
        }
    }
}

// =====================================================================
extern "C" void kernel_launch(void* const* d_in, const int* in_sizes, int n_in,
                              void* d_out, int out_size, void* d_ws, size_t ws_size,
                              hipStream_t stream) {
    (void)in_sizes; (void)n_in; (void)out_size; (void)ws_size;
    const int*   sent  = (const int*)d_in[0];
    const float* emb   = (const float*)d_in[2];
    const float* W_ih0 = (const float*)d_in[3];
    const float* b_ih0 = (const float*)d_in[5];
    const float* b_hh0 = (const float*)d_in[6];
    const float* W_ih1 = (const float*)d_in[7];
    const float* b_ih1 = (const float*)d_in[9];
    const float* b_hh1 = (const float*)d_in[10];
    const float* W_lin = (const float*)d_in[11];
    const float* b_lin = (const float*)d_in[12];
    const float* trans = (const float*)d_in[13];

    char* ws = (char*)d_ws;
    __bf16*        Bp0    = (__bf16*)(ws + OFF_BP0);
    __bf16*        Bp1    = (__bf16*)(ws + OFF_BP1);
    float*         bias0  = (float*)(ws + OFF_BIAS0);
    float*         bias1  = (float*)(ws + OFF_BIAS1);
    float*         logits = (float*)(ws + OFF_LOGITS);
    float*         Pws    = (float*)(ws + OFF_P);
    float*         fvs    = (float*)(ws + OFF_FVS);
    unsigned char* bpws   = (unsigned char*)(ws + OFF_BP);
    unsigned char* Fws    = (unsigned char*)(ws + OFF_F);
    int*           entry  = (int*)(ws + OFF_ENTRY);
    int*           bestws = (int*)(ws + OFF_BEST);
    float*         out    = (float*)d_out;

    pack_w0  <<<2048, 256, 0, stream>>>(W_ih0, Bp0);
    pack_w1  <<<4096, 256, 0, stream>>>(W_ih1, Bp1);
    prep_bias<<<8,    256, 0, stream>>>(b_ih0, b_hh0, b_ih1, b_hh1, bias0, bias1);

    fused_lstm<<<L_ / 16, 256, 0, stream>>>(sent, emb, Bp0, Bp1, bias0, bias1,
                                            W_lin, b_lin, logits);

    viterbi_chunk_prod<<<64, 32, 0, stream>>>(logits, trans, Pws);
    viterbi_boundary  <<<1,  32, 0, stream>>>(Pws, trans, fvs, out, bestws);
    viterbi_replay    <<<64, 32, 0, stream>>>(logits, trans, fvs, bpws, Fws);
    viterbi_entry     <<<1,  32, 0, stream>>>(Fws, bestws, entry);
    viterbi_backtrace <<<64, 32, 0, stream>>>(bpws, entry, out);
}
